// RGCN_58557584113860
// MI455X (gfx1250) — matrix operand order repaired and checked by reading, not compiled, not run on hardware
//
#include <hip/hip_runtime.h>
#include <hip/hip_bf16.h>

// ---------------------------------------------------------------------------
// RGCN (3 layers) + weighted-sum pooling + MLP head for MI455X (gfx1250).
// Per-relation GEMMs run on v_wmma_f32_16x16x32_f16 (wave32 WMMA).
// A-tiles are staged through LDS (normalized f32 -> f16) so the dense
// aggregated tensor is read from HBM exactly once per chunk.
// ---------------------------------------------------------------------------

typedef _Float16 v16h  __attribute__((ext_vector_type(16)));
typedef _Float16 half8 __attribute__((ext_vector_type(8)));
typedef _Float16 half4 __attribute__((ext_vector_type(4)));
typedef float    v8f   __attribute__((ext_vector_type(8)));
typedef float    f4    __attribute__((ext_vector_type(4)));

#define RNUM 65     // relations
#define RC   5      // relations per chunk (65 = 13 * 5)
#define NCHUNK 13
#define HDIM 256    // hidden width (H), also max F
#define MDIM 64     // MLP width
#define APAD 8      // LDS row padding (halves) -> conflict-free b128 reads

// ------------------------------- utility kernels ---------------------------

__global__ void zero_f32_kernel(float* __restrict__ p, int n) {
    int i = blockIdx.x * blockDim.x + threadIdx.x;
    if (i < n) p[i] = 0.f;
}

__global__ void cvt_f32_f16_kernel(const float* __restrict__ s, _Float16* __restrict__ d, int n) {
    int i = blockIdx.x * blockDim.x + threadIdx.x;
    if (i < n) d[i] = (_Float16)s[i];
}

__global__ void relu_cvt_kernel(const float* __restrict__ s, _Float16* __restrict__ d, int n) {
    int i = blockIdx.x * blockDim.x + threadIdx.x;
    if (i < n) d[i] = (_Float16)fmaxf(s[i], 0.f);
}

__global__ void count_kernel(const int* __restrict__ etype, const int* __restrict__ eindex,
                             float* __restrict__ cnt, int E, int N) {
    int e = blockIdx.x * blockDim.x + threadIdx.x;
    if (e >= E) return;
    int t = etype[e];
    int d = eindex[E + e];
    atomicAdd(&cnt[(size_t)t * N + d], 1.f);
}

__global__ void inv_kernel(const float* __restrict__ cnt, float* __restrict__ inv, int n) {
    int i = blockIdx.x * blockDim.x + threadIdx.x;
    if (i < n) inv[i] = 1.f / fmaxf(cnt[i], 1.f);
}

// Repack W[r][k][h] (f32, row-major) -> Wt[r][h][k] (f16) so each B fragment
// is two contiguous 16B loads per lane.
__global__ void pack_w_kernel(const float* __restrict__ W, _Float16* __restrict__ Wt, int F) {
    int i = blockIdx.x * blockDim.x + threadIdx.x;
    int tot = RNUM * F * HDIM;
    if (i >= tot) return;
    int r = i / (F * HDIM);
    int rem = i - r * (F * HDIM);
    int k = rem / HDIM;
    int h = rem - k * HDIM;
    Wt[((size_t)r * HDIM + h) * F + k] = (_Float16)W[i];
}

__global__ void pack_root_kernel(const float* __restrict__ Rm, _Float16* __restrict__ Rt, int F) {
    int i = blockIdx.x * blockDim.x + threadIdx.x;
    if (i >= F * HDIM) return;
    int k = i / HDIM;
    int h = i - k * HDIM;
    Rt[(size_t)h * F + k] = (_Float16)Rm[i];
}

// Edge scatter: S[rel_local][dst][k..k+7] += h_in[src][k..k+7]  (fp32 atomics)
__global__ void scatter_kernel(const int* __restrict__ eindex, const int* __restrict__ etype,
                               const _Float16* __restrict__ hin, float* __restrict__ S,
                               int E, int N, int F, int r0) {
    int perEdge = F >> 3;
    int tid = blockIdx.x * blockDim.x + threadIdx.x;
    int e = tid / perEdge;
    if (e >= E) return;
    int t = etype[e];
    int rl = t - r0;
    if (rl < 0 || rl >= RC) return;
    int k = (tid - e * perEdge) << 3;
    int s = eindex[e];
    int d = eindex[E + e];
    half8 v = *(const half8*)(hin + (size_t)s * F + k);
    float* dp = S + ((size_t)rl * N + d) * F + k;
#pragma unroll
    for (int j = 0; j < 8; ++j) atomicAdd(dp + j, (float)v[j]);
}

// ------------------------------- WMMA GEMM ---------------------------------
// One 128-thread block (4 waves) computes a 16(M) x 256(full H) slab of
// out[N][H]. The A tile (16 x F) is staged once into LDS (normalized,
// f32 -> f16); wave w covers columns [64w, 64w+64).
// ROOT mode:  out = h16 @ root^T + bias                      (init)
// REL  mode:  out += sum_{r in chunk} (S_r * inv_r) @ W_r^T  (accumulate)
template <bool ROOT>
__global__ void rgcn_gemm_kernel(const float* __restrict__ S, const _Float16* __restrict__ A16,
                                 const _Float16* __restrict__ Wt, const float* __restrict__ invc,
                                 const float* __restrict__ bias, float* __restrict__ outb,
                                 int N, int F, int r0) {
    __shared__ _Float16 Ash[16 * (HDIM + APAD)];
    const int P   = F + APAD;          // LDS row pitch (halves)
    const int tid = threadIdx.x;       // 0..127
    const int lane = tid & 31;
    const int wave = tid >> 5;         // 0..3 -> h-slab
    const int hi  = lane >> 4;         // half-wave selector
    const int r16 = lane & 15;
    const int n0  = blockIdx.x * 16;
    const int h0  = wave * 64;

    v8f c0 = {}, c1 = {}, c2 = {}, c3 = {};
    const int nrel = ROOT ? 1 : RC;

    for (int rl = 0; rl < nrel; ++rl) {
        // ---- stage A tile (16 x F) into LDS ----
        if (ROOT) {
            const int iters = (16 * F) / (8 * 128);   // F=256:4, F=64:1
#pragma unroll
            for (int it = 0; it < iters; ++it) {
                int i = (it * 128 + tid) * 8;
                int row = i / F, k = i - row * F;
                half8 v = *(const half8*)(A16 + (size_t)(n0 + row) * F + k);
                *(half8*)(Ash + row * P + k) = v;
            }
        } else {
            const int iters = (16 * F) / (4 * 128);   // F=256:8, F=64:2
#pragma unroll
            for (int it = 0; it < iters; ++it) {
                int i = (it * 128 + tid) * 4;
                int row = i / F, k = i - row * F;
                float sc = invc[(size_t)(r0 + rl) * N + n0 + row];
                f4 v = *(const f4*)(S + ((size_t)rl * N + n0 + row) * F + k);
                half4 hv;
#pragma unroll
                for (int j = 0; j < 4; ++j) hv[j] = (_Float16)(v[j] * sc);
                *(half4*)(Ash + row * P + k) = hv;
            }
        }
        __syncthreads();

        // ---- K loop: A fragments from LDS, B fragments from L2-hot Wt ----
        for (int k0 = 0; k0 < F; k0 += 32) {
            // 16-bit A 16x32 layout (matches B): lanes 0-15 hold K=k0..k0+15;
            // lanes 16-31 hold K=k0+16..k0+31.
            half8 lo = *(const half8*)(Ash + r16 * P + k0 + hi * 16);
            half8 up = *(const half8*)(Ash + r16 * P + k0 + hi * 16 + 8);
            v16h a;
#pragma unroll
            for (int j = 0; j < 8; ++j) { a[j] = lo[j]; a[8 + j] = up[j]; }
#pragma unroll
            for (int nt = 0; nt < 4; ++nt) {
                size_t rbase = ROOT ? 0 : (size_t)(r0 + rl) * HDIM;
                const _Float16* bp = Wt + (rbase + h0 + nt * 16 + r16) * F + k0 + hi * 16;
                half8 b0 = *(const half8*)bp;
                half8 b1 = *(const half8*)(bp + 8);
                v16h b;
#pragma unroll
                for (int j = 0; j < 8; ++j) { b[j] = b0[j]; b[8 + j] = b1[j]; }
                v8f& cc = (nt == 0) ? c0 : (nt == 1) ? c1 : (nt == 2) ? c2 : c3;
                cc = __builtin_amdgcn_wmma_f32_16x16x32_f16(
                        false, a, false, b, (short)0, cc, false, false);
            }
        }
        __syncthreads();   // A tile may be restaged next relation
    }

    // Epilogue: C/D layout — lane l: col = l&15; rows = (l>>4)*8 + j.
    const int rowbase = n0 + hi * 8;
#pragma unroll
    for (int nt = 0; nt < 4; ++nt) {
        v8f cc = (nt == 0) ? c0 : (nt == 1) ? c1 : (nt == 2) ? c2 : c3;
        int hcol = h0 + nt * 16 + r16;
#pragma unroll
        for (int j = 0; j < 8; ++j) {
            size_t idx = (size_t)(rowbase + j) * HDIM + hcol;
            if (ROOT) outb[idx] = cc[j] + bias[hcol];
            else      outb[idx] += cc[j];
        }
    }
}

// ------------------------------- pooling + MLP -----------------------------

__global__ void pool_kernel(const _Float16* __restrict__ h3, const int* __restrict__ batch,
                            const float* __restrict__ ws_w, const float* __restrict__ ws_b,
                            float* __restrict__ pooled, int N) {
    __shared__ float red[32];
    int n = blockIdx.x, t = threadIdx.x;
    float hv[8];
    float s = 0.f;
#pragma unroll
    for (int j = 0; j < 8; ++j) {
        int k = t + 32 * j;
        hv[j] = (float)h3[(size_t)n * HDIM + k];
        s += hv[j] * ws_w[k];
    }
    red[t] = s;
    __syncthreads();
    if (t == 0) {
        float tot = 0.f;
        for (int i = 0; i < 32; ++i) tot += red[i];
        red[0] = 1.f / (1.f + __expf(-(tot + ws_b[0])));
    }
    __syncthreads();
    float gate = red[0];
    int b = batch[n];
#pragma unroll
    for (int j = 0; j < 8; ++j)
        atomicAdd(&pooled[(size_t)b * HDIM + t + 32 * j], gate * hv[j]);
}

__global__ void mlp_kernel(const float* __restrict__ pooled,
                           const float* __restrict__ w1, const float* __restrict__ b1,
                           const float* __restrict__ w2, const float* __restrict__ b2,
                           const float* __restrict__ w3, const float* __restrict__ b3,
                           float* __restrict__ out) {
    __shared__ float sp[HDIM];
    __shared__ float s1[MDIM];
    __shared__ float s2[MDIM];
    int b = blockIdx.x, t = threadIdx.x;
    for (int i = t; i < HDIM; i += MDIM) sp[i] = pooled[(size_t)b * HDIM + i];
    __syncthreads();
    float acc = b1[t];
    for (int k = 0; k < HDIM; ++k) acc += sp[k] * w1[k * MDIM + t];
    s1[t] = fmaxf(acc, 0.f);
    __syncthreads();
    acc = b2[t];
    for (int k = 0; k < MDIM; ++k) acc += s1[k] * w2[k * MDIM + t];
    s2[t] = fmaxf(acc, 0.f);
    __syncthreads();
    if (t == 0) {
        float s = b3[0];
        for (int k = 0; k < MDIM; ++k) s += s2[k] * w3[k];
        out[b] = s;
    }
}

// ------------------------------- host driver -------------------------------

static inline int cdiv(long long a, long long b) { return (int)((a + b - 1) / b); }

extern "C" void kernel_launch(void* const* d_in, const int* in_sizes, int n_in,
                              void* d_out, int out_size, void* d_ws, size_t ws_size,
                              hipStream_t stream) {
    const float* x       = (const float*)d_in[0];
    const int*   eindex  = (const int*)d_in[1];
    const int*   etype   = (const int*)d_in[2];
    const int*   batch   = (const int*)d_in[3];
    const float* Wl[3]   = { (const float*)d_in[4], (const float*)d_in[7], (const float*)d_in[10] };
    const float* rootl[3]= { (const float*)d_in[5], (const float*)d_in[8], (const float*)d_in[11] };
    const float* biasl[3]= { (const float*)d_in[6], (const float*)d_in[9], (const float*)d_in[12] };
    const float* ws_w = (const float*)d_in[13];
    const float* ws_b = (const float*)d_in[14];
    const float* m_w1 = (const float*)d_in[15];
    const float* m_b1 = (const float*)d_in[16];
    const float* m_w2 = (const float*)d_in[17];
    const float* m_b2 = (const float*)d_in[18];
    const float* m_w3 = (const float*)d_in[19];
    const float* m_b3 = (const float*)d_in[20];

    const int N  = in_sizes[3];            // 8000
    const int E  = in_sizes[2];            // 192000
    const int F0 = in_sizes[0] / N;        // 64
    const int B  = out_size;               // 128

    // Workspace bump allocator (256B aligned). Total ~70 MB.
    char* wp = (char*)d_ws;
    auto balloc = [&](size_t bytes) -> void* {
        void* p = (void*)wp;
        wp += (bytes + 255) & ~(size_t)255;
        return p;
    };
    _Float16* hA   = (_Float16*)balloc((size_t)N * HDIM * 2);
    _Float16* hB   = (_Float16*)balloc((size_t)N * HDIM * 2);
    float*    cnt  = (float*)balloc((size_t)RNUM * N * 4);
    float*    invc = (float*)balloc((size_t)RNUM * N * 4);
    _Float16* Wt   = (_Float16*)balloc((size_t)RNUM * HDIM * HDIM * 2);
    _Float16* Rt   = (_Float16*)balloc((size_t)HDIM * HDIM * 2);
    float*    S    = (float*)balloc((size_t)RC * N * HDIM * 4);
    float*    outb = (float*)balloc((size_t)N * HDIM * 4);
    float*    pooled = (float*)balloc((size_t)B * HDIM * 4);

    const int T = 256;

    // Per-(relation,dst) edge counts -> inverse (layer-independent).
    zero_f32_kernel<<<cdiv((long long)RNUM * N, T), T, 0, stream>>>(cnt, RNUM * N);
    count_kernel<<<cdiv(E, T), T, 0, stream>>>(etype, eindex, cnt, E, N);
    inv_kernel<<<cdiv((long long)RNUM * N, T), T, 0, stream>>>(cnt, invc, RNUM * N);

    // x -> f16 into hA.
    cvt_f32_f16_kernel<<<cdiv((long long)N * F0, T), T, 0, stream>>>(x, hA, N * F0);

    for (int l = 0; l < 3; ++l) {
        const int F = (l == 0) ? F0 : HDIM;
        const _Float16* hin  = (l == 1) ? hB : hA;   // l0: hA(x16), l1: hB(h1), l2: hA(h2)
        _Float16*       hout = (l == 1) ? hA : hB;   // l0->hB, l1->hA, l2->hB

        pack_w_kernel<<<cdiv((long long)RNUM * F * HDIM, T), T, 0, stream>>>(Wl[l], Wt, F);
        pack_root_kernel<<<cdiv((long long)F * HDIM, T), T, 0, stream>>>(rootl[l], Rt, F);

        // out = h @ root + bias   (WMMA, init)
        rgcn_gemm_kernel<true><<<N / 16, 128, 0, stream>>>(
            nullptr, hin, Rt, nullptr, biasl[l], outb, N, F, 0);

        // out += sum_r mean_r @ W_r, 13 chunks of 5 relations.
        for (int c = 0; c < NCHUNK; ++c) {
            int r0 = c * RC;
            zero_f32_kernel<<<cdiv((long long)RC * N * F, T), T, 0, stream>>>(S, RC * N * F);
            scatter_kernel<<<cdiv((long long)E * (F / 8), T), T, 0, stream>>>(
                eindex, etype, hin, S, E, N, F, r0);
            rgcn_gemm_kernel<false><<<N / 16, 128, 0, stream>>>(
                S, nullptr, Wt, invc, nullptr, outb, N, F, r0);
        }

        relu_cvt_kernel<<<cdiv((long long)N * HDIM, T), T, 0, stream>>>(outb, hout, N * HDIM);
    }

    // Weighted-sum pooling over graphs, then MLP head.
    zero_f32_kernel<<<cdiv((long long)B * HDIM, T), T, 0, stream>>>(pooled, B * HDIM);
    pool_kernel<<<N, 32, 0, stream>>>(hB, batch, ws_w, ws_b, pooled, N);
    mlp_kernel<<<B, MDIM, 0, stream>>>(pooled, m_w1, m_b1, m_w2, m_b2, m_w3, m_b3, (float*)d_out);
}
